// MultiHeadSelfAttention_74646531605063
// MI455X (gfx1250) — matrix operand order repaired and checked by reading, not compiled, not run on hardware
//
#include <hip/hip_runtime.h>
#include <math.h>

// MI455X / gfx1250: wave32, WMMA bf16 16x16x32 with f32 accumulation.
// Whole pipeline is bf16-resident (one upfront f32->bf16 convert), f32 output.

typedef __attribute__((ext_vector_type(16))) __bf16 v16bf;
typedef __attribute__((ext_vector_type(8)))  __bf16 v8bf;
typedef __attribute__((ext_vector_type(8)))  float  v8f;

#define D_MODEL   1024
#define NUM_HEADS 16
#define HEAD_DIM  64
#define BATCH     2048
#define SROW      2052   // padded LDS row stride (floats)

// ---------------------------------------------------------------------------
// WMMA helper: D = A(16x32 bf16) x B(32x16 bf16) + C(16x16 f32)
// ---------------------------------------------------------------------------
__device__ __forceinline__ v8f wmma_bf16(v16bf a, v16bf b, v8f c) {
  return __builtin_amdgcn_wmma_f32_16x16x32_bf16(
      /*neg_a=*/false, a, /*neg_b=*/false, b,
      /*c_mod=*/(short)0, c, /*reuse_a=*/false, /*reuse_b=*/false);
}

// A-fragment (16x32 bf16, row-major). Lane L: m = L%16, half = L/16.
// elems[0..15] = K = half*16 + 0..15 (contiguous, matches B fragments).
// base = &A[m*lda + k0]; both 16B loads are 16B-aligned (k0 % 32 == 0).
__device__ __forceinline__ v16bf load_a_bf(const __bf16* __restrict__ base,
                                           int half) {
  v8bf lo = *(const v8bf*)(base + half * 16);
  v8bf hi = *(const v8bf*)(base + half * 16 + 8);
  return __builtin_shufflevector(lo, hi, 0, 1, 2, 3, 4, 5, 6, 7,
                                 8, 9, 10, 11, 12, 13, 14, 15);
}

// B-fragment (32x16) with contiguous column: lane L holds 16 consecutive K of
// row n of Bt (Bt stored N x K). One aligned 32-byte load.
// base = &Bt[n*ldb + k0 + half*16]

// ---------------------------------------------------------------------------
// f32 -> bf16 bulk convert (n multiple of 8)
// ---------------------------------------------------------------------------
__global__ __launch_bounds__(256)
void cvt_f32_to_bf16(const float* __restrict__ in, __bf16* __restrict__ out,
                     int n) {
  const int i = (blockIdx.x * 256 + threadIdx.x) * 8;
  if (i >= n) return;
  float4 a = *(const float4*)(in + i);
  float4 b = *(const float4*)(in + i + 4);
  v8bf o;
  o[0] = (__bf16)a.x; o[1] = (__bf16)a.y; o[2] = (__bf16)a.z; o[3] = (__bf16)a.w;
  o[4] = (__bf16)b.x; o[5] = (__bf16)b.y; o[6] = (__bf16)b.z; o[7] = (__bf16)b.w;
  *(v8bf*)(out + i) = o;
}

// ---------------------------------------------------------------------------
// C = A(MxK) @ Bt^T (+ bias), A/Bt bf16 row-major, f32 accum.
// Block tile 64x256, 8 waves (2x4), wave tile 32x64 (2x4 WMMA tiles), K-step 32.
// TRANS_OUT: write C transposed (N x M) as bf16 with vectorized 16B stores.
// ---------------------------------------------------------------------------
template <typename OutT, bool TRANS_OUT>
__global__ __launch_bounds__(256)
void gemm_bf16(const __bf16* __restrict__ A, const __bf16* __restrict__ Bt,
               const float* __restrict__ bias, OutT* __restrict__ C,
               int M, int N, int K) {
  const int tid  = threadIdx.x;
  const int lane = tid & 31;
  const int wv   = tid >> 5;       // 0..7
  const int wm   = wv & 1;         // 2 wave rows
  const int wn   = wv >> 1;        // 4 wave cols
  const int m0   = blockIdx.y * 64 + wm * 32;
  const int n0   = blockIdx.x * 256 + wn * 64;
  const int half = lane >> 4;
  const int l16  = lane & 15;

  v8f acc[2][4] = {};

  const __bf16* Arow0 = A + (size_t)(m0 + l16) * K;
  const __bf16* Arow1 = A + (size_t)(m0 + 16 + l16) * K;
  const __bf16* Brow[4];
#pragma unroll
  for (int ns = 0; ns < 4; ++ns)
    Brow[ns] = Bt + (size_t)(n0 + ns * 16 + l16) * K + half * 16;

  for (int k0 = 0; k0 < K; k0 += 32) {
    v16bf a0 = load_a_bf(Arow0 + k0, half);
    v16bf a1 = load_a_bf(Arow1 + k0, half);
    v16bf b[4];
#pragma unroll
    for (int ns = 0; ns < 4; ++ns) b[ns] = *(const v16bf*)(Brow[ns] + k0);
#pragma unroll
    for (int ns = 0; ns < 4; ++ns) {
      acc[0][ns] = wmma_bf16(a0, b[ns], acc[0][ns]);
      acc[1][ns] = wmma_bf16(a1, b[ns], acc[1][ns]);
    }
  }

#pragma unroll
  for (int ms = 0; ms < 2; ++ms)
#pragma unroll
    for (int ns = 0; ns < 4; ++ns) {
      const int n = n0 + ns * 16 + l16;
      const float bn = bias ? bias[n] : 0.0f;
      if constexpr (TRANS_OUT) {
        // C is N x M bf16; 8 consecutive m per lane -> one 16B store
        const int mb = m0 + ms * 16 + half * 8;
        v8bf o;
#pragma unroll
        for (int r = 0; r < 8; ++r) o[r] = (__bf16)(acc[ms][ns][r] + bn);
        *(v8bf*)((__bf16*)C + (size_t)n * M + mb) = o;
      } else {
#pragma unroll
        for (int r = 0; r < 8; ++r) {
          const int m = m0 + ms * 16 + r + half * 8;
          C[(size_t)m * N + n] = (OutT)(acc[ms][ns][r] + bn);
        }
      }
    }
}

// ---------------------------------------------------------------------------
// Attention: 16-row block, loop over 16 heads; WMMA score stripe (16x2048),
// LDS softmax, accumulate softmax/16; write bf16 A_mean rows.
// Dynamic LDS: scores + accum (16 x SROW each) + red[256] ~ 263 KB
// (legal on CDNA5: 320 KB LDS per workgroup; one WG per WGP).
// ---------------------------------------------------------------------------
__global__ __launch_bounds__(256)
void attn_mean_kernel(const __bf16* __restrict__ qp,
                      const __bf16* __restrict__ kp,
                      __bf16* __restrict__ Amean) {
  extern __shared__ float lds[];
  float* scores = lds;                // 16 x SROW
  float* accum  = lds + 16 * SROW;    // 16 x SROW
  float* red    = lds + 32 * SROW;    // 256 floats

  const int tid  = threadIdx.x;
  const int lane = tid & 31;
  const int wv   = tid >> 5;
  const int half = lane >> 4;
  const int l16  = lane & 15;
  const int r0   = blockIdx.x * 16;

  for (int i = tid; i < 16 * SROW; i += 256) accum[i] = 0.0f;
  __syncthreads();

  for (int h = 0; h < NUM_HEADS; ++h) {
    const __bf16* qbase = qp + (size_t)(r0 + l16) * D_MODEL + h * HEAD_DIM;
    v16bf qf0 = load_a_bf(qbase, half);
    v16bf qf1 = load_a_bf(qbase + 32, half);

    for (int ct = wv; ct < BATCH / 16; ct += 8) {
      const int c0 = ct * 16;
      const __bf16* kbase =
          kp + (size_t)(c0 + l16) * D_MODEL + h * HEAD_DIM + half * 16;
      v16bf kf0 = *(const v16bf*)(kbase);
      v16bf kf1 = *(const v16bf*)(kbase + 32);
      v8f s = {};
      s = wmma_bf16(qf0, kf0, s);
      s = wmma_bf16(qf1, kf1, s);
#pragma unroll
      for (int r = 0; r < 8; ++r)
        scores[(size_t)(r + half * 8) * SROW + c0 + l16] = s[r] * 0.125f;
    }
    __syncthreads();

    const int row = tid >> 4;
    const int sub = tid & 15;
    float* srow = scores + (size_t)row * SROW;

    float mx = -3.4e38f;
    for (int c = sub; c < BATCH; c += 16) mx = fmaxf(mx, srow[c]);
    red[tid] = mx;
    __syncthreads();
    float rmax = -3.4e38f;
#pragma unroll
    for (int i = 0; i < 16; ++i) rmax = fmaxf(rmax, red[row * 16 + i]);
    __syncthreads();

    float ssum = 0.0f;
    for (int c = sub; c < BATCH; c += 16) {
      float e = __expf(srow[c] - rmax);
      srow[c] = e;
      ssum += e;
    }
    red[tid] = ssum;
    __syncthreads();
    float tot = 0.0f;
#pragma unroll
    for (int i = 0; i < 16; ++i) tot += red[row * 16 + i];
    const float inv = 1.0f / (tot * (float)NUM_HEADS);

    float* arow = accum + (size_t)row * SROW;
    for (int c = sub; c < BATCH; c += 16) arow[c] += srow[c] * inv;
    __syncthreads();
  }

  for (int i = tid; i < 16 * BATCH; i += 256) {
    const int rr = i >> 11;
    const int cc = i & (BATCH - 1);
    Amean[(size_t)(r0 + rr) * BATCH + cc] = (__bf16)accum[(size_t)rr * SROW + cc];
  }
}

// ---------------------------------------------------------------------------
// Workspace layout (bytes, needs 44 MB total):
//  [ 0, 4M)  qbf      [ 4, 8M)  kbf      [ 8,12M)  vbf
//  [12,14M)  Wqbf     [14,16M)  Wkbf     [16,18M)  Wvbf    [18,20M)  Wobf
//  [20,24M)  qp(bf16) [24,28M)  kp(bf16) [28,32M)  vpT(bf16, 1024x2048)
//  [32,40M)  Amean(bf16, 2048x2048)      [40,44M)  Opre(bf16)
// ---------------------------------------------------------------------------
extern "C" void kernel_launch(void* const* d_in, const int* in_sizes, int n_in,
                              void* d_out, int out_size, void* d_ws, size_t ws_size,
                              hipStream_t stream) {
  (void)in_sizes; (void)n_in; (void)out_size; (void)ws_size;
  const float* q  = (const float*)d_in[0];
  const float* k  = (const float*)d_in[1];
  const float* v  = (const float*)d_in[2];
  const float* Wq = (const float*)d_in[3];
  const float* bq = (const float*)d_in[4];
  const float* Wk = (const float*)d_in[5];
  const float* bk = (const float*)d_in[6];
  const float* Wv = (const float*)d_in[7];
  const float* bv = (const float*)d_in[8];
  const float* Wo = (const float*)d_in[9];
  const float* bo = (const float*)d_in[10];
  float* out = (float*)d_out;

  char* ws = (char*)d_ws;
  __bf16* qbf   = (__bf16*)(ws + ((size_t)0  << 20));
  __bf16* kbf   = (__bf16*)(ws + ((size_t)4  << 20));
  __bf16* vbf   = (__bf16*)(ws + ((size_t)8  << 20));
  __bf16* Wqbf  = (__bf16*)(ws + ((size_t)12 << 20));
  __bf16* Wkbf  = (__bf16*)(ws + ((size_t)14 << 20));
  __bf16* Wvbf  = (__bf16*)(ws + ((size_t)16 << 20));
  __bf16* Wobf  = (__bf16*)(ws + ((size_t)18 << 20));
  __bf16* qp    = (__bf16*)(ws + ((size_t)20 << 20));
  __bf16* kp    = (__bf16*)(ws + ((size_t)24 << 20));
  __bf16* vpT   = (__bf16*)(ws + ((size_t)28 << 20));  // 1024 x 2048
  __bf16* Amean = (__bf16*)(ws + ((size_t)32 << 20));  // 2048 x 2048
  __bf16* Opre  = (__bf16*)(ws + ((size_t)40 << 20));  // 2048 x 1024

  const dim3 blk(256);
  const int nQKV = BATCH * D_MODEL;    // 2M elems
  const int nW   = D_MODEL * D_MODEL;  // 1M elems

  // 0) one-time f32 -> bf16 conversion of all operands
  cvt_f32_to_bf16<<<dim3(nQKV / 8 / 256), blk, 0, stream>>>(q, qbf, nQKV);
  cvt_f32_to_bf16<<<dim3(nQKV / 8 / 256), blk, 0, stream>>>(k, kbf, nQKV);
  cvt_f32_to_bf16<<<dim3(nQKV / 8 / 256), blk, 0, stream>>>(v, vbf, nQKV);
  cvt_f32_to_bf16<<<dim3(nW / 8 / 256), blk, 0, stream>>>(Wq, Wqbf, nW);
  cvt_f32_to_bf16<<<dim3(nW / 8 / 256), blk, 0, stream>>>(Wk, Wkbf, nW);
  cvt_f32_to_bf16<<<dim3(nW / 8 / 256), blk, 0, stream>>>(Wv, Wvbf, nW);
  cvt_f32_to_bf16<<<dim3(nW / 8 / 256), blk, 0, stream>>>(Wo, Wobf, nW);

  const dim3 gGemm(D_MODEL / 256, BATCH / 64);  // (4, 32), N==1024 for all GEMMs

  // 1) projections (vp written transposed so step 3 gets contiguous B rows)
  gemm_bf16<__bf16, false><<<gGemm, blk, 0, stream>>>(
      qbf, Wqbf, bq, qp, BATCH, D_MODEL, D_MODEL);
  gemm_bf16<__bf16, false><<<gGemm, blk, 0, stream>>>(
      kbf, Wkbf, bk, kp, BATCH, D_MODEL, D_MODEL);
  gemm_bf16<__bf16, true><<<gGemm, blk, 0, stream>>>(
      vbf, Wvbf, bv, vpT, BATCH, D_MODEL, D_MODEL);

  // 2) A_mean = mean_h softmax(Qh Kh^T / 8)
  const size_t smem = (size_t)(32 * SROW + 256) * sizeof(float);  // ~263 KB
  attn_mean_kernel<<<dim3(BATCH / 16), blk, smem, stream>>>(qp, kp, Amean);

  // 3) Opre = A_mean @ vpT^T   (2048x2048 @ 2048x1024)
  gemm_bf16<__bf16, false><<<gGemm, blk, 0, stream>>>(
      Amean, vpT, (const float*)nullptr, Opre, BATCH, D_MODEL, BATCH);

  // 4) out = Opre @ Wo^T + bo  (f32 output)
  gemm_bf16<float, false><<<gGemm, blk, 0, stream>>>(
      Opre, Wobf, bo, out, BATCH, D_MODEL, D_MODEL);
}
